// GPT_SlidingWindow_35003983462690
// MI455X (gfx1250) — matrix operand order repaired and checked by reading, not compiled, not run on hardware
//
#include <hip/hip_runtime.h>

#define L_SEQ   2048
#define DMODEL  1024
#define NLAYERS 8
#define NHEADS  16
#define NKVH    8
#define HDIM    64
#define VOCAB   32000
#define HID     2730
#define HIDP    2816   // HID padded to multiple of 128 (block tile N)
#define WINSZ   128

#define QBLK    64     // queries per attention block
#define KROWS   192    // QBLK + WINSZ staged K/V rows
#define KSTR    68     // padded LDS row stride in floats (272B, 16B aligned)

typedef unsigned short u16;
typedef __attribute__((ext_vector_type(16))) __bf16 bf16x16;
typedef __attribute__((ext_vector_type(8)))  float  f32x8;

union FragAB { bf16x16 v; uint4 q[2]; };

__device__ __forceinline__ u16 f2bf(float f) {
  unsigned int u = __float_as_uint(f);
  u += 0x7FFFu + ((u >> 16) & 1u);          // round-to-nearest-even
  return (u16)(u >> 16);
}

// ---------------------------------------------------------------------------
// C[M,Np] (f32) = A[M,Kp] (bf16, row-major) @ Bt[Np,Kp] (bf16, [N,K] layout)
// Optional fused residual R (f32, stride Np). Requires M%128==0, Np%128==0,
// Kp%32==0. Block = 256 threads = 8 waves (2M x 4N); wave tile 64x32
// (4x2 WMMA tiles, 8 v_wmma per 32-wide K step).
// ---------------------------------------------------------------------------
__global__ __launch_bounds__(256) void gemm_bf16(
    const u16* __restrict__ A, const u16* __restrict__ Bt,
    const float* __restrict__ R, float* __restrict__ C,
    int M, int Np, int Kp)
{
  const int lane = threadIdx.x & 31;
  const int wave = threadIdx.x >> 5;     // 0..7
  const int half = lane >> 4;            // 0: lanes 0-15, 1: lanes 16-31
  const int l16  = lane & 15;

  const int m_wave = blockIdx.y * 128 + (wave >> 2) * 64;
  const int n_wave = blockIdx.x * 128 + (wave & 3) * 32;

  f32x8 zero = {};
  f32x8 acc[4][2];
  #pragma unroll
  for (int i = 0; i < 4; ++i) { acc[i][0] = zero; acc[i][1] = zero; }

  const u16* arow[4];
  const u16* brow[2];
  #pragma unroll
  for (int i = 0; i < 4; ++i) arow[i] = A + (size_t)(m_wave + i * 16 + l16) * Kp;
  #pragma unroll
  for (int j = 0; j < 2; ++j) brow[j] = Bt + (size_t)(n_wave + j * 16 + l16) * Kp;

  for (int k0 = 0; k0 < Kp; k0 += 32) {
    FragAB a[4], b[2];
    // A fragment (16x32 bf16): lane half selects K sub-block per ISA layout
    #pragma unroll
    for (int i = 0; i < 4; ++i) {
      a[i].q[0] = *(const uint4*)(arow[i] + k0 + half * 8);
      a[i].q[1] = *(const uint4*)(arow[i] + k0 + 16 + half * 8);
    }
    // B fragment (32x16 bf16): lanes 0-15 K=0..7,16..23, lanes 16-31 K=8..15,24..31
    #pragma unroll
    for (int j = 0; j < 2; ++j) {
      b[j].q[0] = *(const uint4*)(brow[j] + k0 + half * 8);
      b[j].q[1] = *(const uint4*)(brow[j] + k0 + 16 + half * 8);
    }
    #pragma unroll
    for (int i = 0; i < 4; ++i)
      #pragma unroll
      for (int j = 0; j < 2; ++j)
        acc[i][j] = __builtin_amdgcn_wmma_f32_16x16x32_bf16(
            false, a[i].v, false, b[j].v, (short)0, acc[i][j], false, false);
  }

  // C/D layout: VGPR r -> M = r + 8*half ; N = lane&15
  #pragma unroll
  for (int i = 0; i < 4; ++i)
    #pragma unroll
    for (int j = 0; j < 2; ++j) {
      int mb = m_wave + i * 16 + half * 8;
      int nb = n_wave + j * 16 + l16;
      #pragma unroll
      for (int r = 0; r < 8; ++r) {
        size_t idx = (size_t)(mb + r) * Np + nb;
        float v = acc[i][j][r];
        if (R) v += R[idx];
        C[idx] = v;
      }
    }
}

// f32 [K,N] weight -> bf16 [Np,Kp] transposed, zero padded
__global__ void wtrans(const float* __restrict__ in, u16* __restrict__ out,
                       int K, int N, int Kp, int Np)
{
  size_t id = (size_t)blockIdx.x * blockDim.x + threadIdx.x;
  size_t tot = (size_t)Kp * Np;
  if (id >= tot) return;
  int k = (int)(id % Kp);
  int n = (int)(id / Kp);
  float v = (k < K && n < N) ? in[(size_t)k * N + n] : 0.f;
  out[id] = f2bf(v);
}

__global__ void conv_bf16(const float* __restrict__ in, u16* __restrict__ out, size_t n)
{
  size_t id = (size_t)blockIdx.x * blockDim.x + threadIdx.x;
  if (id < n) out[id] = f2bf(in[id]);
}

__global__ void embed(const int* __restrict__ x, const float* __restrict__ tok,
                      float* __restrict__ h)
{
  size_t id = (size_t)blockIdx.x * blockDim.x + threadIdx.x;
  if (id >= (size_t)L_SEQ * DMODEL) return;
  int l = (int)(id / DMODEL);
  int d = (int)(id % DMODEL);
  h[id] = tok[(size_t)x[l] * DMODEL + d];
}

__global__ __launch_bounds__(256) void rmsnorm(const float* __restrict__ h,
    const float* __restrict__ w, u16* __restrict__ out)
{
  __shared__ float red[256];
  int row = blockIdx.x;
  const float* hr = h + (size_t)row * DMODEL;
  float ss = 0.f;
  for (int d = threadIdx.x; d < DMODEL; d += 256) { float x = hr[d]; ss += x * x; }
  red[threadIdx.x] = ss; __syncthreads();
  for (int s = 128; s > 0; s >>= 1) {
    if ((int)threadIdx.x < s) red[threadIdx.x] += red[threadIdx.x + s];
    __syncthreads();
  }
  float inv = rsqrtf(red[0] / (float)DMODEL + 1e-6f);
  for (int d = threadIdx.x; d < DMODEL; d += 256)
    out[(size_t)row * DMODEL + d] = f2bf(hr[d] * inv * w[d]);
}

// in-place RoPE on t[L, nh, 64]
__global__ void rope(float* __restrict__ t, int nh)
{
  int id = blockIdx.x * blockDim.x + threadIdx.x;
  int total = L_SEQ * nh * 32;
  if (id >= total) return;
  int d  = id & 31;
  int hh = (id >> 5) % nh;
  int l  = id / (32 * nh);
  float inv = __expf(-((float)(2 * d) / 64.f) * 9.210340371976184f); // ln(10000)
  float ang = (float)l * inv;
  float c = cosf(ang), s = sinf(ang);
  float* p = t + ((size_t)l * nh + hh) * HDIM;
  float x0 = p[d], x1 = p[d + 32];
  p[d]      = x0 * c - x1 * s;
  p[d + 32] = x1 * c + x0 * s;
}

// ---------------------------------------------------------------------------
// Sliding-window GQA attention. One block per (kv-head, 64-query tile).
// K/V windows (192 rows x 64 f32) staged global->LDS with CDNA5 async-to-LDS
// DMA (ASYNCcnt), then 128 threads (64 queries x 2 GQA heads) run online
// softmax entirely out of LDS. Rows padded to 68 floats: bank = 4*r mod 64,
// so 16 consecutive query rows hit 16 distinct bank groups.
// ---------------------------------------------------------------------------
__global__ __launch_bounds__(128) void attention(const float* __restrict__ Q,
    const float* __restrict__ K, const float* __restrict__ Vv,
    u16* __restrict__ O)
{
  __shared__ float sk[KROWS * KSTR];
  __shared__ float sv[KROWS * KSTR];
  const int kvh = blockIdx.x;          // 0..NKVH-1
  const int q0  = blockIdx.y * QBLK;
  const int tid = threadIdx.x;
  const int j0  = q0 - WINSZ;          // first staged row (may be negative)

  // stage K and V: 192 rows x 16 pieces of 16B each
  for (int c = tid; c < KROWS * 16; c += 128) {
    int r = c >> 4, piece = c & 15;
    int j = j0 + r;
    if (j >= 0) {
      const float* gk = K  + ((size_t)j * NKVH + kvh) * HDIM + piece * 4;
      const float* gv = Vv + ((size_t)j * NKVH + kvh) * HDIM + piece * 4;
      unsigned int lk = (unsigned int)(uintptr_t)(sk + r * KSTR + piece * 4);
      unsigned int lv = (unsigned int)(uintptr_t)(sv + r * KSTR + piece * 4);
      asm volatile("global_load_async_to_lds_b128 %0, %1, off"
                   :: "v"(lk), "v"(gk) : "memory");
      asm volatile("global_load_async_to_lds_b128 %0, %1, off"
                   :: "v"(lv), "v"(gv) : "memory");
    }
  }
  asm volatile("s_wait_asynccnt 0x0" ::: "memory");
  __syncthreads();

  const int i  = q0 + (tid >> 1);          // query index
  const int hh = kvh * 2 + (tid & 1);      // full head index (NREP=2)
  const float* q = Q + ((size_t)i * NHEADS + hh) * HDIM;
  float qv[HDIM], o[HDIM];
  #pragma unroll
  for (int d = 0; d < HDIM; ++d) { qv[d] = q[d]; o[d] = 0.f; }

  float m = -1e30f, ssum = 0.f;
  int jlo = i - WINSZ; if (jlo < 0) jlo = 0;
  for (int j = jlo; j <= i; ++j) {
    const float* kr = sk + (j - j0) * KSTR;
    float s = 0.f;
    #pragma unroll
    for (int d = 0; d < HDIM; ++d) s += qv[d] * kr[d];
    s *= 0.125f;                           // 64^-0.5
    float nm = fmaxf(m, s);
    float corr = __expf(m - nm);
    float p = __expf(s - nm);
    ssum = ssum * corr + p;
    const float* vr = sv + (j - j0) * KSTR;
    #pragma unroll
    for (int d = 0; d < HDIM; ++d) o[d] = o[d] * corr + p * vr[d];
    m = nm;
  }
  float invs = 1.f / ssum;
  u16* op = O + ((size_t)i * NHEADS + hh) * HDIM;
  #pragma unroll
  for (int d = 0; d < HDIM; ++d) op[d] = f2bf(o[d] * invs);
}

__global__ void swiglu(const float* __restrict__ u, const float* __restrict__ g,
                       u16* __restrict__ s, size_t n)
{
  size_t id = (size_t)blockIdx.x * blockDim.x + threadIdx.x;
  if (id >= n) return;
  float x = u[id];
  float si = x / (1.f + __expf(-x));
  s[id] = f2bf(si * g[id]);
}

// ---------------------------------------------------------------------------
extern "C" void kernel_launch(void* const* d_in, const int* in_sizes, int n_in,
                              void* d_out, int out_size, void* d_ws, size_t ws_size,
                              hipStream_t stream)
{
  const int*   x    = (const int*)  d_in[0];
  const float* tok  = (const float*)d_in[1];
  const float* ln1  = (const float*)d_in[2];
  const float* wq   = (const float*)d_in[3];
  const float* wk   = (const float*)d_in[4];
  const float* wv   = (const float*)d_in[5];
  const float* wo   = (const float*)d_in[6];
  const float* ln2  = (const float*)d_in[7];
  const float* w1   = (const float*)d_in[8];
  const float* w2   = (const float*)d_in[9];
  const float* w3   = (const float*)d_in[10];
  const float* lnf  = (const float*)d_in[11];
  float* out = (float*)d_out;

  char* p = (char*)d_ws;
  auto alloc = [&](size_t bytes) -> char* {
    char* r = p; p += (bytes + 255) & ~(size_t)255; return r;
  };

  const size_t KV = NKVH * HDIM;  // 512

  u16* emb_b = (u16*)alloc((size_t)VOCAB * DMODEL * 2);
  u16* wq_t  = (u16*)alloc((size_t)NLAYERS * DMODEL * DMODEL * 2);
  u16* wk_t  = (u16*)alloc((size_t)NLAYERS * KV * DMODEL * 2);
  u16* wv_t  = (u16*)alloc((size_t)NLAYERS * KV * DMODEL * 2);
  u16* wo_t  = (u16*)alloc((size_t)NLAYERS * DMODEL * DMODEL * 2);
  u16* w1_t  = (u16*)alloc((size_t)NLAYERS * HIDP * DMODEL * 2);
  u16* w3_t  = (u16*)alloc((size_t)NLAYERS * HIDP * DMODEL * 2);
  u16* w2_t  = (u16*)alloc((size_t)NLAYERS * DMODEL * HIDP * 2);

  float* h    = (float*)alloc((size_t)L_SEQ * DMODEL * 4);
  u16*   hn   = (u16*)  alloc((size_t)L_SEQ * DMODEL * 2);
  float* qb   = (float*)alloc((size_t)L_SEQ * DMODEL * 4);
  float* kb   = (float*)alloc((size_t)L_SEQ * KV * 4);
  float* vb   = (float*)alloc((size_t)L_SEQ * KV * 4);
  u16*   ab   = (u16*)  alloc((size_t)L_SEQ * DMODEL * 2);
  float* ub   = (float*)alloc((size_t)L_SEQ * HIDP * 4);
  float* gb   = (float*)alloc((size_t)L_SEQ * HIDP * 4);
  u16*   sb   = (u16*)  alloc((size_t)L_SEQ * HIDP * 2);

  auto blocks = [](size_t n, int b) { return (unsigned)((n + b - 1) / b); };

  // --- prep: bf16 embedding (also serves as Bt for tied logits GEMM) ---
  conv_bf16<<<blocks((size_t)VOCAB * DMODEL, 256), 256, 0, stream>>>(
      tok, emb_b, (size_t)VOCAB * DMODEL);

  // --- prep: transpose all weights to bf16 [Np,Kp] ---
  for (int l = 0; l < NLAYERS; ++l) {
    wtrans<<<blocks((size_t)DMODEL * DMODEL, 256), 256, 0, stream>>>(
        wq + (size_t)l * DMODEL * DMODEL, wq_t + (size_t)l * DMODEL * DMODEL,
        DMODEL, DMODEL, DMODEL, DMODEL);
    wtrans<<<blocks((size_t)KV * DMODEL, 256), 256, 0, stream>>>(
        wk + (size_t)l * DMODEL * KV, wk_t + (size_t)l * KV * DMODEL,
        DMODEL, (int)KV, DMODEL, (int)KV);
    wtrans<<<blocks((size_t)KV * DMODEL, 256), 256, 0, stream>>>(
        wv + (size_t)l * DMODEL * KV, wv_t + (size_t)l * KV * DMODEL,
        DMODEL, (int)KV, DMODEL, (int)KV);
    wtrans<<<blocks((size_t)DMODEL * DMODEL, 256), 256, 0, stream>>>(
        wo + (size_t)l * DMODEL * DMODEL, wo_t + (size_t)l * DMODEL * DMODEL,
        DMODEL, DMODEL, DMODEL, DMODEL);
    wtrans<<<blocks((size_t)HIDP * DMODEL, 256), 256, 0, stream>>>(
        w1 + (size_t)l * DMODEL * HID, w1_t + (size_t)l * HIDP * DMODEL,
        DMODEL, HID, DMODEL, HIDP);
    wtrans<<<blocks((size_t)HIDP * DMODEL, 256), 256, 0, stream>>>(
        w3 + (size_t)l * DMODEL * HID, w3_t + (size_t)l * HIDP * DMODEL,
        DMODEL, HID, DMODEL, HIDP);
    wtrans<<<blocks((size_t)DMODEL * HIDP, 256), 256, 0, stream>>>(
        w2 + (size_t)l * HID * DMODEL, w2_t + (size_t)l * DMODEL * HIDP,
        HID, DMODEL, HIDP, DMODEL);
  }

  // --- embedding gather ---
  embed<<<blocks((size_t)L_SEQ * DMODEL, 256), 256, 0, stream>>>(x, tok, h);

  dim3 gB(256);
  for (int l = 0; l < NLAYERS; ++l) {
    rmsnorm<<<L_SEQ, 256, 0, stream>>>(h, ln1 + (size_t)l * DMODEL, hn);

    gemm_bf16<<<dim3(DMODEL / 128, L_SEQ / 128), gB, 0, stream>>>(
        hn, wq_t + (size_t)l * DMODEL * DMODEL, nullptr, qb, L_SEQ, DMODEL, DMODEL);
    gemm_bf16<<<dim3(KV / 128, L_SEQ / 128), gB, 0, stream>>>(
        hn, wk_t + (size_t)l * KV * DMODEL, nullptr, kb, L_SEQ, (int)KV, DMODEL);
    gemm_bf16<<<dim3(KV / 128, L_SEQ / 128), gB, 0, stream>>>(
        hn, wv_t + (size_t)l * KV * DMODEL, nullptr, vb, L_SEQ, (int)KV, DMODEL);

    rope<<<blocks((size_t)L_SEQ * NHEADS * 32, 256), 256, 0, stream>>>(qb, NHEADS);
    rope<<<blocks((size_t)L_SEQ * NKVH * 32, 256), 256, 0, stream>>>(kb, NKVH);

    attention<<<dim3(NKVH, L_SEQ / QBLK), 128, 0, stream>>>(qb, kb, vb, ab);

    gemm_bf16<<<dim3(DMODEL / 128, L_SEQ / 128), gB, 0, stream>>>(
        ab, wo_t + (size_t)l * DMODEL * DMODEL, h, h, L_SEQ, DMODEL, DMODEL);

    rmsnorm<<<L_SEQ, 256, 0, stream>>>(h, ln2 + (size_t)l * DMODEL, hn);

    gemm_bf16<<<dim3(HIDP / 128, L_SEQ / 128), gB, 0, stream>>>(
        hn, w1_t + (size_t)l * HIDP * DMODEL, nullptr, ub, L_SEQ, HIDP, DMODEL);
    gemm_bf16<<<dim3(HIDP / 128, L_SEQ / 128), gB, 0, stream>>>(
        hn, w3_t + (size_t)l * HIDP * DMODEL, nullptr, gb, L_SEQ, HIDP, DMODEL);

    swiglu<<<blocks((size_t)L_SEQ * HIDP, 256), 256, 0, stream>>>(
        ub, gb, sb, (size_t)L_SEQ * HIDP);

    gemm_bf16<<<dim3(DMODEL / 128, L_SEQ / 128), gB, 0, stream>>>(
        sb, w2_t + (size_t)l * DMODEL * HIDP, h, h, L_SEQ, DMODEL, HIDP);
  }

  rmsnorm<<<L_SEQ, 256, 0, stream>>>(h, lnf, hn);

  // logits: [2048,32000] = hn @ tok_emb.T  (emb_b is already [N=V, K=D] bf16)
  gemm_bf16<<<dim3(VOCAB / 128, L_SEQ / 128), gB, 0, stream>>>(
      hn, emb_b, nullptr, out, L_SEQ, VOCAB, DMODEL);
}
